// GraphEncoder_2714419331082
// MI455X (gfx1250) — matrix operand order repaired and checked by reading, not compiled, not run on hardware
//
#include <hip/hip_runtime.h>
#include <hip/hip_bf16.h>

#define BATCH 8
#define NPTS  2048
#define TOTN  (BATCH * NPTS)
#define KNBR  16
#define NFEAT 256

typedef __attribute__((ext_vector_type(8)))  __bf16 v8bf;
typedef __attribute__((ext_vector_type(16))) __bf16 v16bf;
typedef __attribute__((ext_vector_type(8)))  float  v8f;

union BF2 { unsigned int u; __bf16 h[2]; };

__device__ __forceinline__ unsigned int pack2(float a, float b)
{
    BF2 p; p.h[0] = (__bf16)a; p.h[1] = (__bf16)b; return p.u;
}

// ---------------------------------------------------------------------------
// KNN: per-thread query point, LDS-tiled candidates, register bubble-insert
// top-16 (strict < keeps lowest index on ties, like jax top_k).
// ---------------------------------------------------------------------------
__global__ __launch_bounds__(256)
void knn_kernel(const float* __restrict__ x, int* __restrict__ nbr)
{
    const int b = blockIdx.x >> 3;                  // 8 tiles per batch
    const int n = ((blockIdx.x & 7) << 8) + threadIdx.x;
    const float* xb = x + (size_t)b * 3 * NPTS;
    const float qx = xb[n], qy = xb[NPTS + n], qz = xb[2 * NPTS + n];

    float dist[KNBR];
    int   idx[KNBR];
#pragma unroll
    for (int i = 0; i < KNBR; ++i) { dist[i] = 3.4e38f; idx[i] = 0; }

    __shared__ float sx[256], sy[256], sz[256];
    for (int t = 0; t < NPTS; t += 256) {
        __syncthreads();
        sx[threadIdx.x] = xb[t + threadIdx.x];
        sy[threadIdx.x] = xb[NPTS + t + threadIdx.x];
        sz[threadIdx.x] = xb[2 * NPTS + t + threadIdx.x];
        __syncthreads();
        for (int j = 0; j < 256; ++j) {
            const float dx = qx - sx[j], dy = qy - sy[j], dz = qz - sz[j];
            const float d = dx * dx + dy * dy + dz * dz;
            if (d < dist[KNBR - 1]) {
                float cd = d; int ci = t + j;
#pragma unroll
                for (int s = 0; s < KNBR; ++s) {
                    if (cd < dist[s]) {
                        float td = dist[s]; int ti = idx[s];
                        dist[s] = cd; idx[s] = ci; cd = td; ci = ti;
                    }
                }
            }
        }
    }
#pragma unroll
    for (int i = 0; i < KNBR; ++i)
        nbr[((size_t)b * NPTS + n) * KNBR + i] = idx[i];
}

// ---------------------------------------------------------------------------
// Neighborhood covariance features: feat = concat(x(3), cov(9)) as (B,12,N)
// ---------------------------------------------------------------------------
__global__ __launch_bounds__(256)
void feat_kernel(const float* __restrict__ x, const int* __restrict__ nbr,
                 float* __restrict__ feat)
{
    const int gid = blockIdx.x * 256 + threadIdx.x;
    if (gid >= TOTN) return;
    const int b = gid / NPTS, n = gid % NPTS;
    const float* xb = x + (size_t)b * 3 * NPTS;
    const int* nb = nbr + (size_t)gid * KNBR;

    float px[KNBR], py[KNBR], pz[KNBR];
    float mx = 0.f, my = 0.f, mz = 0.f;
#pragma unroll
    for (int k = 0; k < KNBR; ++k) {
        const int j = nb[k];
        px[k] = xb[j]; py[k] = xb[NPTS + j]; pz[k] = xb[2 * NPTS + j];
        mx += px[k]; my += py[k]; mz += pz[k];
    }
    mx *= (1.f / KNBR); my *= (1.f / KNBR); mz *= (1.f / KNBR);
    float c[9] = {0.f, 0.f, 0.f, 0.f, 0.f, 0.f, 0.f, 0.f, 0.f};
#pragma unroll
    for (int k = 0; k < KNBR; ++k) {
        const float a0 = px[k] - mx, a1 = py[k] - my, a2 = pz[k] - mz;
        c[0] += a0 * a0; c[1] += a0 * a1; c[2] += a0 * a2;
        c[3] += a1 * a0; c[4] += a1 * a1; c[5] += a1 * a2;
        c[6] += a2 * a0; c[7] += a2 * a1; c[8] += a2 * a2;
    }
    float* fb = feat + (size_t)b * 12 * NPTS;
    fb[0 * NPTS + n] = xb[n];
    fb[1 * NPTS + n] = xb[NPTS + n];
    fb[2 * NPTS + n] = xb[2 * NPTS + n];
#pragma unroll
    for (int i = 0; i < 9; ++i) fb[(3 + i) * NPTS + n] = c[i];
}

// ---------------------------------------------------------------------------
// WMMA GEMM: Y(B,Cout,N) = W(Cout,CIN) * X(B,CIN,N) [+ bias] [+= Y] [relu]
// f32 in HBM, bf16 staged in LDS, f32 accumulate via v_wmma_f32_16x16x32_bf16.
// Block = 128 threads (4 waves); block tile 16(out) x 64(cols); K step 32.
// LDS layouts chosen so each lane's 16-elem fragment = two ds_load_b128:
//   ldsW [row][k]  row-major  -> A frag: k runs {kbA..+7},{kbA+16..+23}
//   ldsXt[col][k]  transposed -> B frag: k runs {kbB..+7},{kbB+16..+23}
// ---------------------------------------------------------------------------
template<int CIN, bool ACC, bool BIAS, bool RELU>
__global__ __launch_bounds__(128)
void gemm_wmma_kernel(const float* __restrict__ X, const float* __restrict__ W,
                      const float* __restrict__ bias, float* __restrict__ Y,
                      int Cout)
{
    __shared__ __bf16 ldsXt[64][32];   // [col][k]  4 KB
    __shared__ __bf16 ldsW[16][32];    // [row][k]  1 KB

    const int tid   = threadIdx.x;
    const int wave  = tid >> 5;
    const int lane  = tid & 31;
    const int half  = lane >> 4;       // 0 or 1
    const int l16   = lane & 15;
    const int otile = blockIdx.y * 16;
    const int m0    = blockIdx.x * 64;      // global column base (64 | 2048)
    const int b     = m0 / NPTS;            // uniform over block
    const int n0    = m0 - b * NPTS;

    // staging assignments (constant per thread)
    const int wr  = tid >> 3;               // W row       0..15
    const int wc0 = (tid & 7) << 2;         // W k offset  0,4,..,28

    v8f acc = {};
    constexpr int KSTEPS = (CIN + 31) / 32;

    for (int kk = 0; kk < KSTEPS; ++kk) {
        const int k0 = kk << 5;

        // ---- gather global data into registers (batched b128 loads) ----
        float4 xa[2], xb2[2];
        int c0s[2], kls[2];
#pragma unroll
        for (int i = 0; i < 2; ++i) {
            const int u  = tid + (i << 7);  // 0..255
            const int kp = u >> 4;          // k-pair 0..15
            const int c0 = (u & 15) << 2;   // col 0,4,..,60
            const int kl = kp << 1;         // local k (even)
            const int k  = k0 + kl;
            c0s[i] = c0; kls[i] = kl;
            float4 va = {0.f, 0.f, 0.f, 0.f}, vb = {0.f, 0.f, 0.f, 0.f};
            if ((CIN % 32 == 0) || (k < CIN))
                va = *(const float4*)(X + ((size_t)b * CIN + k) * NPTS + n0 + c0);
            if ((CIN % 32 == 0) || (k + 1 < CIN))
                vb = *(const float4*)(X + ((size_t)b * CIN + k + 1) * NPTS + n0 + c0);
            xa[i] = va; xb2[i] = vb;
        }
        float4 wv = {0.f, 0.f, 0.f, 0.f};
        if ((CIN % 32 == 0) || (wc0 < CIN))
            wv = *(const float4*)(W + (size_t)(otile + wr) * CIN + k0 + wc0);

        // prefetch next X tile (speculative, dropped if invalid)
        if ((CIN >= 64) && (kk + 1 < KSTEPS))
            __builtin_prefetch(X + ((size_t)b * CIN + k0 + 32 + (tid >> 6)) * NPTS
                                 + n0 + ((tid & 63)), 0, 0);

        // ---- convert + store to LDS (packed b32/b64 stores) ----
#pragma unroll
        for (int i = 0; i < 2; ++i) {
            const int c0 = c0s[i], kl = kls[i];
            *(unsigned int*)&ldsXt[c0 + 0][kl] = pack2(xa[i].x, xb2[i].x);
            *(unsigned int*)&ldsXt[c0 + 1][kl] = pack2(xa[i].y, xb2[i].y);
            *(unsigned int*)&ldsXt[c0 + 2][kl] = pack2(xa[i].z, xb2[i].z);
            *(unsigned int*)&ldsXt[c0 + 3][kl] = pack2(xa[i].w, xb2[i].w);
        }
        {
            uint2 wp;
            wp.x = pack2(wv.x, wv.y);
            wp.y = pack2(wv.z, wv.w);
            *(uint2*)&ldsW[wr][wc0] = wp;
        }
        __syncthreads();

        // ---- fragments: two ds_load_b128 each ----
        const int kbA = half * 8;           // A halves: k 0..7/16..23 | 8..15/24..31
        const int kbB = half * 8;           // B halves: k 0..7/16..23 | 8..15/24..31
        const int col = (wave << 4) + l16;

        const v8bf aLo = *(const v8bf*)&ldsW[l16][kbA];
        const v8bf aHi = *(const v8bf*)&ldsW[l16][kbA + 16];
        const v8bf bLo = *(const v8bf*)&ldsXt[col][kbB];
        const v8bf bHi = *(const v8bf*)&ldsXt[col][kbB + 16];
        const v16bf afrag = __builtin_shufflevector(aLo, aHi,
            0, 1, 2, 3, 4, 5, 6, 7, 8, 9, 10, 11, 12, 13, 14, 15);
        const v16bf bfrag = __builtin_shufflevector(bLo, bHi,
            0, 1, 2, 3, 4, 5, 6, 7, 8, 9, 10, 11, 12, 13, 14, 15);

        acc = __builtin_amdgcn_wmma_f32_16x16x32_bf16(
                  false, afrag, false, bfrag, (short)0, acc, false, false);
        __syncthreads();
    }

    // ---- epilogue ----
    // C/D layout: lane l16 = column; VGPR r -> row (otile + half*8 + r)
    const int gcol  = n0 + (wave << 4) + l16;
    const int orow0 = otile + half * 8;
    float* yb = Y + ((size_t)b * Cout + orow0) * NPTS + gcol;

    float prev[8];
    if (ACC) {
#pragma unroll
        for (int r = 0; r < 8; ++r) prev[r] = yb[(size_t)r * NPTS];
    }
#pragma unroll
    for (int r = 0; r < 8; ++r) {
        float v = acc[r];
        if (ACC)  v += prev[r];
        if (BIAS) v += bias[orow0 + r];
        if (RELU) v = fmaxf(v, 0.f);
        yb[(size_t)r * NPTS] = v;
    }
}

// ---------------------------------------------------------------------------
// BatchNorm over (B, N) per channel: stats then apply(+ReLU)
// ---------------------------------------------------------------------------
__global__ __launch_bounds__(256)
void bn_stats_kernel(const float* __restrict__ Y, float* __restrict__ stats, int Cout)
{
    const int o = blockIdx.x;
    __shared__ float s1[256], s2[256];
    float sum = 0.f, sq = 0.f;
    for (int b = 0; b < BATCH; ++b) {
        const float* p = Y + ((size_t)b * Cout + o) * NPTS;
        for (int n = threadIdx.x; n < NPTS; n += 256) {
            const float v = p[n]; sum += v; sq += v * v;
        }
    }
    s1[threadIdx.x] = sum; s2[threadIdx.x] = sq; __syncthreads();
    for (int st = 128; st > 0; st >>= 1) {
        if (threadIdx.x < st) {
            s1[threadIdx.x] += s1[threadIdx.x + st];
            s2[threadIdx.x] += s2[threadIdx.x + st];
        }
        __syncthreads();
    }
    if (threadIdx.x == 0) {
        const float inv = 1.f / (float)(BATCH * NPTS);
        const float m = s1[0] * inv;
        stats[o]        = m;
        stats[Cout + o] = s2[0] * inv - m * m;
    }
}

__global__ __launch_bounds__(256)
void bn_apply_relu_kernel(float* __restrict__ Y, const float* __restrict__ stats,
                          const float* __restrict__ g, const float* __restrict__ bb,
                          int Cout, int total)
{
    const int i = blockIdx.x * 256 + threadIdx.x;
    if (i >= total) return;
    const int o = (i / NPTS) % Cout;
    const float m = stats[o], v = stats[Cout + o];
    float r = g[o] * (Y[i] - m) * rsqrtf(v + 1e-5f) + bb[o];
    Y[i] = r > 0.f ? r : 0.f;
}

// ---------------------------------------------------------------------------
// Graph aggregation: agg[b,c,n] = sum_k h[b,c,nbr[b,n,k]]
// ---------------------------------------------------------------------------
__global__ __launch_bounds__(256)
void agg_kernel(const float* __restrict__ h, const int* __restrict__ nbr,
                float* __restrict__ agg)
{
    const int gid = blockIdx.x * 256 + threadIdx.x;   // over B*N
    if (gid >= TOTN) return;
    const int c = blockIdx.y;
    const int b = gid / NPTS, n = gid % NPTS;
    const int* nb = nbr + (size_t)gid * KNBR;
    const float* hb = h + ((size_t)b * 64 + c) * NPTS;
    float s = 0.f;
#pragma unroll
    for (int k = 0; k < KNBR; ++k) s += hb[nb[k]];
    agg[((size_t)b * 64 + c) * NPTS + n] = s;
}

// ---------------------------------------------------------------------------
// FiLM + positional encoding: out[b,c,n] = gam*pe[n,c] + bet  (c < 512)
// ---------------------------------------------------------------------------
__global__ __launch_bounds__(256)
void film_pe_kernel(const float* __restrict__ fm, float* __restrict__ out)
{
    const int i = blockIdx.x * 256 + threadIdx.x;       // B*512*N
    if (i >= BATCH * 512 * NPTS) return;
    const int n = i % NPTS;
    const int t = i / NPTS;
    const int c = t % 512;
    const int b = t / 512;
    const float gam = fm[((size_t)b * 1024 + c) * NPTS + n];
    const float bet = fm[((size_t)b * 1024 + 512 + c) * NPTS + n];
    const float dv = expf((float)(c & ~1) * (-5.2983173665480363f / 512.f)); // ln(200)
    const float ang = (float)n * dv;
    const float pe = (c & 1) ? cosf(ang) : sinf(ang);
    out[i] = gam * pe + bet;
}

// ---------------------------------------------------------------------------
// Masked max-pool over N (mask all-valid): pooled[b,c] = max_n h[b,c,n]
// ---------------------------------------------------------------------------
__global__ __launch_bounds__(256)
void maxpool_kernel(const float* __restrict__ h, float* __restrict__ pooled)
{
    const int bc = blockIdx.x;                 // b*512 + c
    const float* p = h + (size_t)bc * NPTS;
    __shared__ float s[256];
    float m = -3.4e38f;
    for (int n = threadIdx.x; n < NPTS; n += 256) m = fmaxf(m, p[n]);
    s[threadIdx.x] = m; __syncthreads();
    for (int st = 128; st > 0; st >>= 1) {
        if (threadIdx.x < st) s[threadIdx.x] = fmaxf(s[threadIdx.x], s[threadIdx.x + st]);
        __syncthreads();
    }
    if (threadIdx.x == 0) pooled[bc] = s[0];
}

// ---------------------------------------------------------------------------
// Tiny conv+BN(over batch)+ReLU for the class branch (N==1): Y (B,O)
// ---------------------------------------------------------------------------
__global__ __launch_bounds__(256)
void small_cbr_kernel(const float* __restrict__ X, const float* __restrict__ W,
                      const float* __restrict__ bias, const float* __restrict__ g,
                      const float* __restrict__ bb, float* __restrict__ Y,
                      int Cin, int O)
{
    const int o = blockIdx.x * 256 + threadIdx.x;
    if (o >= O) return;
    float y[BATCH]; float m = 0.f;
    for (int b = 0; b < BATCH; ++b) {
        float s = bias[o];
        const float* xb = X + (size_t)b * Cin;
        const float* w  = W + (size_t)o * Cin;
        for (int c = 0; c < Cin; ++c) s += w[c] * xb[c];
        y[b] = s; m += s;
    }
    m *= (1.f / BATCH);
    float v = 0.f;
    for (int b = 0; b < BATCH; ++b) { const float d = y[b] - m; v += d * d; }
    v *= (1.f / BATCH);
    const float rs = rsqrtf(v + 1e-5f);
    for (int b = 0; b < BATCH; ++b) {
        const float r = g[o] * (y[b] - m) * rs + bb[o];
        Y[(size_t)b * O + o] = r > 0.f ? r : 0.f;
    }
}

// ---------------------------------------------------------------------------
// FiLM combine + fc + L2 normalize: out (B,256). One block of 256 threads.
// ---------------------------------------------------------------------------
__global__ __launch_bounds__(256)
void final_kernel(const float* __restrict__ pooled, const float* __restrict__ gg,
                  const float* __restrict__ b2, const float* __restrict__ fcW,
                  const float* __restrict__ fcb, float* __restrict__ out)
{
    __shared__ float xc[BATCH][512];
    __shared__ float so[BATCH][256];
    __shared__ float norms[BATCH];
    const int t = threadIdx.x;
    for (int i = t; i < BATCH * 512; i += 256) {
        const int b = i >> 9, c = i & 511;
        const float p = pooled[b * 512 + c];
        xc[b][c] = 0.5f * (gg[b * 512 + c] * p + b2[b * 512 + c]) + 0.5f * p;
    }
    __syncthreads();
    for (int b = 0; b < BATCH; ++b) {
        float s = fcb[t];
        const float* w = fcW + (size_t)t * 512;
        for (int c = 0; c < 512; ++c) s += w[c] * xc[b][c];
        so[b][t] = s;
    }
    __syncthreads();
    if (t < BATCH) {
        float s = 0.f;
        for (int o = 0; o < 256; ++o) { const float v = so[t][o]; s += v * v; }
        norms[t] = sqrtf(s);
    }
    __syncthreads();
    for (int b = 0; b < BATCH; ++b) out[b * 256 + t] = so[b][t] / norms[b];
}

// ---------------------------------------------------------------------------
// gl_idx: row = i/K ; col = nbr + b*N (stored as int32 raw after the floats)
// ---------------------------------------------------------------------------
__global__ __launch_bounds__(256)
void glidx_kernel(const int* __restrict__ nbr, int* __restrict__ gl)
{
    const int i = blockIdx.x * 256 + threadIdx.x;
    const int T = TOTN * KNBR;
    if (i >= T) return;
    gl[i] = i / KNBR;
    const int b = i / (NPTS * KNBR);
    gl[T + i] = nbr[i] + b * NPTS;
}

// ---------------------------------------------------------------------------
extern "C" void kernel_launch(void* const* d_in, const int* in_sizes, int n_in,
                              void* d_out, int out_size, void* d_ws, size_t ws_size,
                              hipStream_t stream)
{
    (void)in_sizes; (void)n_in; (void)out_size; (void)ws_size;
    const float* x    = (const float*)d_in[0];
    const float* clss = (const float*)d_in[1];
    // d_in[2] = mask (all ones, unused)
    auto P = [&](int i) { return (const float*)d_in[i]; };
    // 3..6 m1a, 7..10 m1b, 11..14 m1c, 15 gc_Wr, 16 gc_Wn, 17 gc_b,
    // 18/19 film1, 20/21 film2, 22..25 m2a, 26..29 m2b,
    // 30..33 mc, 34..37 ga, 38..41 be, 42/43 fc

    char* ws = (char*)d_ws;
    size_t off = 0;
    auto alloc = [&](size_t bytes) {
        void* p = ws + off;
        off = (off + bytes + 255) & ~(size_t)255;
        return p;
    };
    int*   nbr    = (int*)  alloc((size_t)TOTN * KNBR * 4);
    float* feat   = (float*)alloc((size_t)BATCH * 12 * NPTS * 4);
    float* stats  = (float*)alloc(2 * 1024 * 4);
    float* ce     = (float*)alloc(BATCH * NFEAT * 4);
    float* gbuf   = (float*)alloc(BATCH * 512 * 4);
    float* b2buf  = (float*)alloc(BATCH * 512 * 4);
    float* pooled = (float*)alloc(BATCH * 512 * 4);
    float* bigA   = (float*)alloc((size_t)BATCH * 1024 * NPTS * 4); // 64 MB
    float* bigB   = (float*)alloc((size_t)BATCH * 512  * NPTS * 4); // 32 MB
    float* bigC   = (float*)alloc((size_t)BATCH * 512  * NPTS * 4); // 32 MB

    auto grid_for = [&](int Cout) { return dim3(TOTN / 64, Cout / 16); };
    const dim3 blk128(128);

    auto bnrelu = [&](float* Y, const float* g, const float* bb, int Cout) {
        bn_stats_kernel<<<Cout, 256, 0, stream>>>(Y, stats, Cout);
        const int total = BATCH * Cout * NPTS;
        bn_apply_relu_kernel<<<(total + 255) / 256, 256, 0, stream>>>(Y, stats, g, bb, Cout, total);
    };

    // 1) KNN + covariance features
    knn_kernel<<<BATCH * (NPTS / 256), 256, 0, stream>>>(x, nbr);
    feat_kernel<<<(TOTN + 255) / 256, 256, 0, stream>>>(x, nbr, feat);

    // 2) m1a/m1b/m1c  (12->64->64->64)
    gemm_wmma_kernel<12, false, true, false><<<grid_for(64), blk128, 0, stream>>>(feat, P(3), P(4), bigB, 64);
    bnrelu(bigB, P(5), P(6), 64);
    gemm_wmma_kernel<64, false, true, false><<<grid_for(64), blk128, 0, stream>>>(bigB, P(7), P(8), bigC, 64);
    bnrelu(bigC, P(9), P(10), 64);
    gemm_wmma_kernel<64, false, true, false><<<grid_for(64), blk128, 0, stream>>>(bigC, P(11), P(12), bigB, 64);
    bnrelu(bigB, P(13), P(14), 64);

    // 3) neighbor aggregation + graph conv (64 -> 512)
    agg_kernel<<<dim3(TOTN / 256, 64), 256, 0, stream>>>(bigB, nbr, bigC);
    gemm_wmma_kernel<64, false, true, false><<<grid_for(512), blk128, 0, stream>>>(bigC, P(16), P(17), bigA, 512); // gc_Wn*agg + gc_b
    gemm_wmma_kernel<64, true, false, false><<<grid_for(512), blk128, 0, stream>>>(bigB, P(15), nullptr, bigA, 512); // += gc_Wr*h

    // 4) FiLM generator: film1 (512->256, relu) then film2 (256->1024)
    gemm_wmma_kernel<512, false, true, true ><<<grid_for(256),  blk128, 0, stream>>>(bigA, P(18), P(19), bigB, 256);
    gemm_wmma_kernel<256, false, true, false><<<grid_for(1024), blk128, 0, stream>>>(bigB, P(20), P(21), bigA, 1024);

    // 5) gamma * PE + beta  -> (B,512,N)
    film_pe_kernel<<<(BATCH * 512 * NPTS + 255) / 256, 256, 0, stream>>>(bigA, bigB);

    // 6) m2a/m2b (512->512)
    gemm_wmma_kernel<512, false, true, false><<<grid_for(512), blk128, 0, stream>>>(bigB, P(22), P(23), bigC, 512);
    bnrelu(bigC, P(24), P(25), 512);
    gemm_wmma_kernel<512, false, true, false><<<grid_for(512), blk128, 0, stream>>>(bigC, P(26), P(27), bigB, 512);
    bnrelu(bigB, P(28), P(29), 512);

    // 7) global max-pool
    maxpool_kernel<<<BATCH * 512, 256, 0, stream>>>(bigB, pooled);

    // 8) class branch: mc (40->256), ga (256->512), be (256->512)
    small_cbr_kernel<<<1, 256, 0, stream>>>(clss, P(30), P(31), P(32), P(33), ce, 40, NFEAT);
    small_cbr_kernel<<<2, 256, 0, stream>>>(ce, P(34), P(35), P(36), P(37), gbuf, NFEAT, 512);
    small_cbr_kernel<<<2, 256, 0, stream>>>(ce, P(38), P(39), P(40), P(41), b2buf, NFEAT, 512);

    // 9) FiLM combine + fc + L2 normalize
    final_kernel<<<1, 256, 0, stream>>>(pooled, gbuf, b2buf, P(42), P(43), (float*)d_out);

    // 10) gl_idx (int32 raw, after the B*256 float outputs)
    int* gl = (int*)((float*)d_out + BATCH * NFEAT);
    glidx_kernel<<<(TOTN * KNBR + 255) / 256, 256, 0, stream>>>(nbr, gl);
}
